// RGAT_18047452578195
// MI455X (gfx1250) — compile-verified, matrix operand order checked by reading, not run on hardware
//
#include <hip/hip_runtime.h>
#include <math.h>

// ---------------------------------------------------------------------------
// RGAT (3-layer, cosine-threshold attention) for MI455X / gfx1250.
//  - GEMMs: v_wmma_f32_16x16x32_bf16; each wave computes a 16x64 strip
//    (4 N-tiles) so every A fragment feeds 4 WMMAs.
//  - Edge gathers / scatters: L2-resident (h, xl, agg all << 192MB L2).
//  - Segment softmax folded: denom = 1 (self loop) + sum(att) via f32 atomics.
//  - f32->bf16 feature conversion fused into the per-node norm pass.
// ---------------------------------------------------------------------------

#define THRV      0.1f
#define ATT_FLOOR 1e-6f
#define COS_EPS   1e-8f
#define LN_EPS    1e-5f

typedef __attribute__((ext_vector_type(16))) __bf16       v16bf;
typedef __attribute__((ext_vector_type(8)))  float        v8f;
typedef __attribute__((ext_vector_type(4)))  unsigned int u32x4;
typedef __attribute__((ext_vector_type(2)))  unsigned int u32x2;
typedef __attribute__((ext_vector_type(4)))  float        f32x4;

union Frag { v16bf v; u32x4 q[2]; };

static __device__ __forceinline__ float wave_sum(float v) {
#pragma unroll
  for (int m = 16; m >= 1; m >>= 1) v += __shfl_xor(v, m, 32);
  return v;
}
static __device__ __forceinline__ float wave_max(float v) {
#pragma unroll
  for (int m = 16; m >= 1; m >>= 1) v = fmaxf(v, __shfl_xor(v, m, 32));
  return v;
}
static __device__ __forceinline__ unsigned int f2bf(float f) {
  unsigned int u = __float_as_uint(f);
  u += 0x7FFFu + ((u >> 16) & 1u);   // round-to-nearest-even
  return u >> 16;
}

// --- zero fill (vectorized, n multiple of 4) -------------------------------
__global__ void k_zero(f32x4* __restrict__ p, long n4) {
  long i = (long)blockIdx.x * blockDim.x + threadIdx.x;
  if (i < n4) p[i] = (f32x4){0.0f, 0.0f, 0.0f, 0.0f};
}

// --- weight f32 [128 x outdim] -> bf16 swizzled into B-fragment order ------
// slot = ((tn*4 + kc)*32 + lane); 16 contiguous bf16 per slot.
__global__ void k_wswz(const float* __restrict__ W, unsigned short* __restrict__ dst, int outdim) {
  int slot = blockIdx.x * blockDim.x + threadIdx.x;
  int total = (outdim >> 4) * 4 * 32;
  if (slot >= total) return;
  int lane = slot & 31;
  int kc   = (slot >> 5) & 3;
  int tn   = slot >> 7;
  int kb   = (lane >> 4) * 8;
  int ncol = tn * 16 + (lane & 15);
#pragma unroll
  for (int e2 = 0; e2 < 16; ++e2) {
    int k = kc * 32 + (e2 < 8 ? kb + e2 : 16 + kb + (e2 - 8));
    dst[(long)slot * 16 + e2] = (unsigned short)f2bf(W[(long)k * outdim + ncol]);
  }
}

// --- per-node norms + denom init + fused f32->bf16 feature conversion ------
__global__ void k_node_norm(const float* __restrict__ h, unsigned short* __restrict__ hb,
                            float* __restrict__ norms, float* __restrict__ denom, int n) {
  int node = (int)(((long)blockIdx.x * blockDim.x + threadIdx.x) >> 5);
  int lane = threadIdx.x & 31;
  if (node >= n) return;
  const f32x4* row = (const f32x4*)(h + (long)node * 128);
  f32x4 v = row[lane];
  // fused bf16 pack: this wave already owns the whole row
  u32x2 pk;
  pk.x = f2bf(v.x) | (f2bf(v.y) << 16);
  pk.y = f2bf(v.z) | (f2bf(v.w) << 16);
  *(u32x2*)(hb + (long)node * 128 + lane * 4) = pk;
  float s = v.x * v.x + v.y * v.y + v.z * v.z + v.w * v.w;
  s = wave_sum(s);
  if (lane == 0) {
    norms[node] = fmaxf(sqrtf(s), COS_EPS);
    denom[node] = 1.0f;                    // self-loop cosine == 1
  }
}

// --- per-edge cosine attention + atomic denom accumulation -----------------
__global__ void k_edge_att(const float* __restrict__ h, const int* __restrict__ src,
                           const int* __restrict__ dst, const float* __restrict__ norms,
                           float* __restrict__ att, float* __restrict__ denom, int e) {
  int edge = (int)(((long)blockIdx.x * blockDim.x + threadIdx.x) >> 5);
  int lane = threadIdx.x & 31;
  if (edge >= e) return;
  int s = src[edge], d = dst[edge];
  const f32x4* rs = (const f32x4*)(h + (long)s * 128);
  const f32x4* rd = (const f32x4*)(h + (long)d * 128);
  f32x4 a4 = rs[lane], b4 = rd[lane];
  float dot = a4.x * b4.x + a4.y * b4.y + a4.z * b4.z + a4.w * b4.w;
  dot = wave_sum(dot);
  if (lane == 0) {
    float a = dot / (norms[s] * norms[d]);
    if (a < THRV) a = ATT_FLOOR;
    att[edge] = a;
    atomicAdd(&denom[d], a);
  }
}

// --- bf16 WMMA GEMM: xl[n x outdim] = h[n x 128] @ W[128 x outdim] ---------
// One wave per 16x64 output strip: A fragment reused across 4 N-tiles.
__global__ void k_gemm(const unsigned short* __restrict__ hb,
                       const unsigned short* __restrict__ wswz,
                       float* __restrict__ xl,
                       int nRows, int nTilesM, int nGroupsN, int outdim) {
  int wv   = (int)(((long)blockIdx.x * blockDim.x + threadIdx.x) >> 5);
  int lane = threadIdx.x & 31;
  if (wv >= nTilesM * nGroupsN) return;          // wave-uniform: EXEC stays all-1s
  int tm  = wv / nGroupsN;
  int tg  = wv - tm * nGroupsN;                  // group of 4 consecutive N tiles
  int row = tm * 16 + (lane & 15);
  if (row >= nRows) row = nRows - 1;             // tail clamp (unused for N=50000)
  int kb  = (lane >> 4) * 8;

  v8f c[4] = {};
  const unsigned short* abase = hb + (long)row * 128 + kb;
  const unsigned short* bbase = wswz + (((long)tg * 16) * 32 + lane) * 16;
#pragma unroll
  for (int kc = 0; kc < 4; ++kc) {
    Frag a;
    a.q[0] = *(const u32x4*)(abase + kc * 32);        // K = kc*32 + kb .. +7
    a.q[1] = *(const u32x4*)(abase + kc * 32 + 16);   // K = kc*32 + 16 + kb .. +7
#pragma unroll
    for (int t = 0; t < 4; ++t) {
      Frag b;
      const unsigned short* bp = bbase + (long)(t * 4 + kc) * 32 * 16;
      b.q[0] = *(const u32x4*)(bp);
      b.q[1] = *(const u32x4*)(bp + 8);
      c[t] = __builtin_amdgcn_wmma_f32_16x16x32_bf16(false, a.v, false, b.v,
                                                     (short)0, c[t], false, false);
    }
  }
  // C layout: VGPR r -> M = r + (lane>=16 ? 8 : 0); N = lane & 15
  int mb   = tm * 16 + ((lane >> 4) * 8);
  bool full = (tm * 16 + 16 <= nRows);           // wave-uniform fast path
#pragma unroll
  for (int t = 0; t < 4; ++t) {
    long base = (long)mb * outdim + tg * 64 + t * 16 + (lane & 15);
    if (full) {
#pragma unroll
      for (int r = 0; r < 8; ++r) xl[base + (long)r * outdim] = c[t][r];
    } else {
#pragma unroll
      for (int r = 0; r < 8; ++r)
        if (mb + r < nRows) xl[base + (long)r * outdim] = c[t][r];
    }
  }
}

// --- edge scatter: agg[dst] += xl[src] * (att / denom[dst]) ----------------
__global__ void k_aggregate(const float* __restrict__ xl, const float* __restrict__ att,
                            const float* __restrict__ denom, const int* __restrict__ src,
                            const int* __restrict__ dst, float* __restrict__ agg,
                            int e, int outdim) {
  int edge = (int)(((long)blockIdx.x * blockDim.x + threadIdx.x) >> 5);
  int lane = threadIdx.x & 31;
  if (edge >= e) return;
  int s = src[edge], d = dst[edge];
  float w = att[edge] / denom[d];
  for (int c = lane; c < outdim; c += 32)
    atomicAdd(&agg[(long)d * outdim + c], xl[(long)s * outdim + c] * w);
}

// --- epilogue (128 cols): + self-msg + bias, LayerNorm, ReLU ---------------
__global__ void k_ln_relu(const float* __restrict__ agg, const float* __restrict__ xl,
                          const float* __restrict__ denom, const float* __restrict__ bias,
                          const float* __restrict__ g, const float* __restrict__ be,
                          float* __restrict__ out, int n) {
  int node = (int)(((long)blockIdx.x * blockDim.x + threadIdx.x) >> 5);
  int lane = threadIdx.x & 31;
  if (node >= n) return;
  float inv = 1.0f / denom[node];
  float v[4], s = 0.0f;
#pragma unroll
  for (int j = 0; j < 4; ++j) {
    int c = lane * 4 + j;
    v[j] = agg[(long)node * 128 + c] + xl[(long)node * 128 + c] * inv + bias[c];
    s += v[j];
  }
  s = wave_sum(s);
  float m = s * (1.0f / 128.0f);
  float q = 0.0f;
#pragma unroll
  for (int j = 0; j < 4; ++j) { float t = v[j] - m; q += t * t; }
  q = wave_sum(q);
  float r = rsqrtf(q * (1.0f / 128.0f) + LN_EPS);
#pragma unroll
  for (int j = 0; j < 4; ++j) {
    int c = lane * 4 + j;
    float o = (v[j] - m) * r * g[c] + be[c];
    out[(long)node * 128 + c] = fmaxf(o, 0.0f);
  }
}

// --- epilogue (64 cols): + self-msg + bias, log_softmax --------------------
__global__ void k_logsoftmax(const float* __restrict__ agg, const float* __restrict__ xl,
                             const float* __restrict__ denom, const float* __restrict__ bias,
                             float* __restrict__ out, int n) {
  int node = (int)(((long)blockIdx.x * blockDim.x + threadIdx.x) >> 5);
  int lane = threadIdx.x & 31;
  if (node >= n) return;
  float inv = 1.0f / denom[node];
  float v[2];
#pragma unroll
  for (int j = 0; j < 2; ++j) {
    int c = lane * 2 + j;
    v[j] = agg[(long)node * 64 + c] + xl[(long)node * 64 + c] * inv + bias[c];
  }
  float mx = wave_max(fmaxf(v[0], v[1]));
  float se = wave_sum(expf(v[0] - mx) + expf(v[1] - mx));
  float lse = mx + logf(se);
#pragma unroll
  for (int j = 0; j < 2; ++j) {
    int c = lane * 2 + j;
    out[(long)node * 64 + c] = v[j] - lse;
  }
}

// ---------------------------------------------------------------------------
static inline int cdiv(long a, long b) { return (int)((a + b - 1) / b); }

extern "C" void kernel_launch(void* const* d_in, const int* in_sizes, int n_in,
                              void* d_out, int out_size, void* d_ws, size_t ws_size,
                              hipStream_t stream) {
  const float* x   = (const float*)d_in[0];
  const int*   ei  = (const int*)d_in[1];
  const float* W0  = (const float*)d_in[2];
  const float* b0  = (const float*)d_in[3];
  const float* W1  = (const float*)d_in[4];
  const float* b1  = (const float*)d_in[5];
  const float* W2  = (const float*)d_in[6];
  const float* b2  = (const float*)d_in[7];
  const float* g1  = (const float*)d_in[8];
  const float* be1 = (const float*)d_in[9];
  const float* g2  = (const float*)d_in[10];
  const float* be2 = (const float*)d_in[11];

  const int n = in_sizes[0] / 128;   // 50000
  const int e = in_sizes[1] / 2;     // 800000
  const int* src = ei;
  const int* dst = ei + e;

  // workspace carve-up (all chunks multiple-of-4 floats -> 16B aligned)
  float* ws = (float*)d_ws;
  long o = 0;
  float* denom = ws + o; o += n;
  float* norms = ws + o; o += n;
  float* att   = ws + o; o += e;
  float* xl    = ws + o; o += (long)n * 128;
  float* agg   = ws + o; o += (long)n * 128;
  float* hA    = ws + o; o += (long)n * 128;
  float* hB    = ws + o; o += (long)n * 128;
  unsigned short* hb  = (unsigned short*)(ws + o); o += (long)n * 64;
  unsigned short* w0b = (unsigned short*)(ws + o); o += (128L * 128) / 2;
  unsigned short* w1b = (unsigned short*)(ws + o); o += (128L * 128) / 2;
  unsigned short* w2b = (unsigned short*)(ws + o); o += (128L * 64) / 2;
  (void)ws_size; (void)n_in;

  // weight conversion + fragment swizzle (once per launch; deterministic)
  k_wswz<<<cdiv(8 * 4 * 32, 128), 128, 0, stream>>>(W0, w0b, 128);
  k_wswz<<<cdiv(8 * 4 * 32, 128), 128, 0, stream>>>(W1, w1b, 128);
  k_wswz<<<cdiv(4 * 4 * 32, 128), 128, 0, stream>>>(W2, w2b, 64);

  const int tilesM = cdiv(n, 16);
  auto conv = [&](const float* hin, const unsigned short* wb, int outdim) {
    k_node_norm<<<cdiv((long)n * 32, 256), 256, 0, stream>>>(hin, hb, norms, denom, n);
    k_zero<<<cdiv((long)n * outdim / 4, 256), 256, 0, stream>>>(
        (f32x4*)agg, (long)n * outdim / 4);
    k_edge_att<<<cdiv((long)e * 32, 256), 256, 0, stream>>>(hin, src, dst, norms, att, denom, e);
    int groupsN = outdim / 64;                   // 16x64 strip per wave
    k_gemm<<<cdiv((long)tilesM * groupsN * 32, 256), 256, 0, stream>>>(
        hb, wb, xl, n, tilesM, groupsN, outdim);
    k_aggregate<<<cdiv((long)e * 32, 256), 256, 0, stream>>>(
        xl, att, denom, src, dst, agg, e, outdim);
  };

  // layer 1: x -> hA
  conv(x, w0b, 128);
  k_ln_relu<<<cdiv((long)n * 32, 256), 256, 0, stream>>>(agg, xl, denom, b0, g1, be1, hA, n);
  // layer 2: hA -> hB
  conv(hA, w1b, 128);
  k_ln_relu<<<cdiv((long)n * 32, 256), 256, 0, stream>>>(agg, xl, denom, b1, g2, be2, hB, n);
  // layer 3: hB -> d_out (log_softmax over 64)
  conv(hB, w2b, 64);
  k_logsoftmax<<<cdiv((long)n * 32, 256), 256, 0, stream>>>(agg, xl, denom, b2, (float*)d_out, n);
}
